// LSTM_Cell_Model_85177791414679
// MI455X (gfx1250) — matrix operand order repaired and checked by reading, not compiled, not run on hardware
//
#include <hip/hip_runtime.h>

// Problem constants (match reference)
#define V_   50000
#define E_   128
#define H_   256
#define C_   2
#define B_   256
#define T_   512
#define G4H  1024            // 4*H
#define KTOT 384             // E + H (fused input+recurrent K)
#define NKT  12              // KTOT/32
#define NNT  64              // G4H/16
#define NBLK 128             // persistent grid: 128 blocks x 2 waves = 256 waves
#define TPB  64              // 2 waves/block -> 1 wave per SIMD at high VGPR use

typedef __attribute__((ext_vector_type(16))) __bf16 bf16x16;
typedef __attribute__((ext_vector_type(8)))  float  f32x8;

union ABReg { bf16x16 v; uint4 q[2]; };   // 32 bytes, two 16B halves

__device__ __forceinline__ unsigned short f32_to_bf16(float f) {
    unsigned u = __float_as_uint(f);
    u += 0x7FFFu + ((u >> 16) & 1u);      // round-to-nearest-even
    return (unsigned short)(u >> 16);
}

__device__ __forceinline__ float sigm(float x) {
    return 1.0f / (1.0f + __expf(-x));
}

// -------- prep kernel 1: emb f32 -> bf16 (one-time 25.6MB -> 12.8MB) --------
__global__ void k_emb_bf16(const float* __restrict__ emb,
                           unsigned short* __restrict__ embb, int n) {
    int i = blockIdx.x * blockDim.x + threadIdx.x;
    int stride = gridDim.x * blockDim.x;
    for (; i < n; i += stride) embb[i] = f32_to_bf16(emb[i]);
}

// -------- prep kernel 2: pack [W_ih^T ; W_hh^T] into WMMA-B lane layout ------
// B tile (kt,nt): lane L element j -> k = kt*32 + (L>=16?8:0) + (j<8 ? j : 8+j),
//                                     n = nt*16 + (L&15)   (ISA 7.12.2, 16-bit B)
// Stored contiguously: Bpack[(kt*NNT+nt)*512 + L*16 + j]  (32B per lane)
__global__ void k_prep(const float* __restrict__ Wih, const float* __restrict__ Whh,
                       unsigned short* __restrict__ Bpack,
                       unsigned short* __restrict__ hbf,
                       float* __restrict__ h_out, float* __restrict__ c_out,
                       unsigned* __restrict__ bar) {
    int i = blockIdx.x * blockDim.x + threadIdx.x;
    int stride = gridDim.x * blockDim.x;
    for (int idx = i; idx < KTOT * G4H; idx += stride) {
        int tile   = idx >> 9;          // /512
        int within = idx & 511;
        int lane   = within >> 4;
        int j      = within & 15;
        int kt     = tile / NNT;
        int nt     = tile % NNT;
        int k      = kt * 32 + ((lane >= 16) ? 8 : 0) + ((j < 8) ? j : 8 + j);
        int n      = nt * 16 + (lane & 15);
        float v = (k < E_) ? Wih[n * E_ + k] : Whh[n * H_ + (k - E_)];
        Bpack[idx] = f32_to_bf16(v);
    }
    for (int idx = i; idx < B_ * H_; idx += stride) {
        hbf[idx]   = 0;       // h0 = 0 (bf16 working copy)
        h_out[idx] = 0.0f;
        c_out[idx] = 0.0f;    // c0 = 0
    }
    if (i == 0) *bar = 0u;    // grid-barrier generation counter
}

// -------- software grid barrier (device scope, never resets within a call) --
__device__ __forceinline__ void grid_barrier(unsigned* bar, unsigned target) {
    __threadfence();
    __syncthreads();
    if (threadIdx.x == 0) {
        __hip_atomic_fetch_add(bar, 1u, __ATOMIC_RELEASE, __HIP_MEMORY_SCOPE_AGENT);
        while (__hip_atomic_load(bar, __ATOMIC_ACQUIRE, __HIP_MEMORY_SCOPE_AGENT) < target) {
            __builtin_amdgcn_s_sleep(1);
        }
    }
    __syncthreads();
    __threadfence();
}

// -------- persistent LSTM kernel: 512 fused steps, one launch ---------------
__global__ void __launch_bounds__(TPB, 1)
k_lstm(const int* __restrict__ x,
       const unsigned short* __restrict__ embb,
       const unsigned short* __restrict__ Bpack,
       const float* __restrict__ b_ih, const float* __restrict__ b_hh,
       const float* __restrict__ W_lin, const float* __restrict__ b_lin,
       unsigned short* __restrict__ hbf,
       float* __restrict__ h_out, float* __restrict__ c_out,
       float* __restrict__ out, unsigned* __restrict__ bar)
{
    const int lane = threadIdx.x & 31;
    const int widx = threadIdx.x >> 5;
    const int w    = blockIdx.x * (TPB / 32) + widx;  // 0..255
    const int mt   = w & 15;       // batch tile
    const int ht   = w >> 4;       // hidden-col tile (shared by block's 2 waves)
    const int hi   = lane >> 4;    // lane half (selects K sub-range, ISA layout)
    const int ml   = lane & 15;

    const int bA   = mt * 16 + ml;          // batch row this lane loads for A
    const int hcol = ht * 16 + ml;          // hidden column this lane owns in C/D

    // Gate biases (b_ih + b_hh) are time-invariant: hoist out of the scan.
    float bias[4];
#pragma unroll
    for (int g = 0; g < 4; ++g) bias[g] = b_ih[g * H_ + hcol] + b_hh[g * H_ + hcol];

    for (int t = 0; t < T_; ++t) {
        // ---- issue ALL A-tile loads up front (24 x global_load_b128 in flight).
        // Recurrent h tiles first: they are the only step-serial dependency and
        // their latency hides under the 16 embedding WMMAs below.
        ABReg ah[8];
#pragma unroll
        for (int kt2 = 0; kt2 < 8; ++kt2) {
            const unsigned short* ap = hbf + bA * H_ + kt2 * 32 + hi * 8;
            ah[kt2].q[0] = *(const uint4*)(ap);        // K = +hi*8 + 0..7
            ah[kt2].q[1] = *(const uint4*)(ap + 16);   // K = +16+hi*8 + 0..7
        }
        const int tok = x[bA * T_ + t];
        ABReg ae[4];
#pragma unroll
        for (int kt = 0; kt < 4; ++kt) {
            const unsigned short* ap = embb + (long)tok * E_ + kt * 32 + hi * 8;
            ae[kt].q[0] = *(const uint4*)(ap);
            ae[kt].q[1] = *(const uint4*)(ap + 16);
        }
        if (t + 1 < T_) {                   // hide next gather under this step
            int tok2 = x[bA * T_ + t + 1];
            __builtin_prefetch(embb + (long)tok2 * E_, 0, 1);   // global_prefetch_b8
        }

        f32x8 acc[4] = {};                  // i, f, g, o accumulators (f32)

        // ---- K = 0..127 : embedding part (B tiles stay VGPR-resident) ----
#pragma unroll
        for (int kt = 0; kt < 4; ++kt) {
#pragma unroll
            for (int g = 0; g < 4; ++g) {
                const bf16x16 bv =
                    *(const bf16x16*)(Bpack + ((kt * NNT + (g * 16 + ht)) * 512 + lane * 16));
                acc[g] = __builtin_amdgcn_wmma_f32_16x16x32_bf16(
                    false, ae[kt].v, false, bv, (short)0, acc[g], false, false);
            }
        }

        // ---- K = 128..383 : recurrent part ----
#pragma unroll
        for (int kt2 = 0; kt2 < 8; ++kt2) {
            const int kt = 4 + kt2;
#pragma unroll
            for (int g = 0; g < 4; ++g) {
                const bf16x16 bv =
                    *(const bf16x16*)(Bpack + ((kt * NNT + (g * 16 + ht)) * 512 + lane * 16));
                acc[g] = __builtin_amdgcn_wmma_f32_16x16x32_bf16(
                    false, ah[kt2].v, false, bv, (short)0, acc[g], false, false);
            }
        }

        // ---- elementwise cell update (wave-local: this wave owns (mt,ht)) ----
        // C/D layout: lane holds N=hcol; VGPR r holds M = mt*16 + hi*8 + r
#pragma unroll
        for (int r = 0; r < 8; ++r) {
            const int m   = mt * 16 + hi * 8 + r;
            const int idx = m * H_ + hcol;
            const float gi = acc[0][r] + bias[0];
            const float gf = acc[1][r] + bias[1];
            const float gg = acc[2][r] + bias[2];
            const float go = acc[3][r] + bias[3];
            const float c_new = sigm(gf) * c_out[idx] + sigm(gi) * tanhf(gg);
            const float h_new = sigm(go) * tanhf(c_new);
            c_out[idx] = c_new;             // f32 (also final output region)
            h_out[idx] = h_new;             // f32 (final output region)
            hbf[idx]   = f32_to_bf16(h_new);// bf16 copy feeds next step's A
        }

        grid_barrier(bar, (unsigned)((t + 1) * NBLK));
    }

    // ---- final tiny linear: out = c @ W_lin^T + b_lin  (256x256x2) ----
    if (blockIdx.x == 0) {
        for (int b = threadIdx.x; b < B_; b += TPB) {
#pragma unroll
            for (int cc = 0; cc < C_; ++cc) {
                float s = b_lin[cc];
                for (int k = 0; k < H_; ++k) s += c_out[b * H_ + k] * W_lin[cc * H_ + k];
                out[b * C_ + cc] = s;
            }
        }
    }
}

extern "C" void kernel_launch(void* const* d_in, const int* in_sizes, int n_in,
                              void* d_out, int out_size, void* d_ws, size_t ws_size,
                              hipStream_t stream) {
    const int*   x     = (const int*)  d_in[0];
    const float* emb   = (const float*)d_in[1];
    const float* W_ih  = (const float*)d_in[2];
    const float* W_hh  = (const float*)d_in[3];
    const float* b_ih  = (const float*)d_in[4];
    const float* b_hh  = (const float*)d_in[5];
    const float* W_lin = (const float*)d_in[6];
    const float* b_lin = (const float*)d_in[7];

    // Workspace layout (all offsets 256B-aligned):
    //   [0)            emb_bf16   : V*E*2       = 12,800,000 B
    //   [12,800,000)   Bpack      : 384*1024*2  =    786,432 B
    //   [13,586,432)   h_bf16     : B*H*2       =    131,072 B
    //   [13,717,504)   barrier    : 4 B
    char* ws = (char*)d_ws;
    unsigned short* embb  = (unsigned short*)(ws);
    unsigned short* Bpack = (unsigned short*)(ws + 12800000);
    unsigned short* hbf   = (unsigned short*)(ws + 12800000 + 786432);
    unsigned*       bar   = (unsigned*)      (ws + 12800000 + 786432 + 131072);

    // d_out = [h (B*H) | c (B*H) | out (B*C)] f32 — h/c state lives in place.
    float* h_out = (float*)d_out;
    float* c_out = h_out + B_ * H_;
    float* outp  = c_out + B_ * H_;

    k_emb_bf16<<<2048, 256, 0, stream>>>(emb, embb, V_ * E_);
    k_prep    <<<1024, 256, 0, stream>>>(W_ih, W_hh, Bpack, hbf, h_out, c_out, bar);
    k_lstm    <<<NBLK, TPB, 0, stream>>>(x, embb, Bpack, b_ih, b_hh, W_lin, b_lin,
                                         hbf, h_out, c_out, outp, bar);
}
